// Cayley_Rot_54443005444547
// MI455X (gfx1250) — matrix operand order repaired and checked by reading, not compiled, not run on hardware
//
#include <hip/hip_runtime.h>
#include <math.h>

#define NUM_CLS 512
#define NUM_DIM 256
#define NGF     32640
#define ITERS   16

typedef __attribute__((ext_vector_type(16))) _Float16 v16h;
typedef __attribute__((ext_vector_type(8)))  _Float16 v8h;
typedef __attribute__((ext_vector_type(8)))  float    v8f;

union U16 { v16h v; v8h h[2]; };

// LDS layout (bytes):
//   Shi : 256*256 f16          @      0  (131072)
//   Slo : 256*256 f16          @ 131072  (131072)
//   xs  : 256 f32              @ 262144  (  1024)
//   yf  : 2*256 f32 (dbl buf)  @ 263168  (  2048)
//   yh  : 2*256 f16 (dbl buf)  @ 265216  (  1024)
//   yl  : 2*256 f16 (dbl buf)  @ 266240  (  1024)
#define SMEM_BYTES 267264

__device__ __forceinline__ int trioff(int i) {
    // elements before row i of the strict upper triangle (row-major triu_indices)
    return i * (NUM_DIM - 1) - (i * (i - 1)) / 2;
}

__global__ void __launch_bounds__(256)
cayley_rot_kernel(const float* __restrict__ x,   // (NUM_DIM, NUM_CLS)
                  const float* __restrict__ W,   // (NUM_CLS, NGF)
                  float* __restrict__ out) {     // (NUM_DIM, NUM_CLS)
    extern __shared__ char smem[];
    _Float16* Shi = (_Float16*)(smem);
    _Float16* Slo = (_Float16*)(smem + 131072);
    float*    xs  = (float*)   (smem + 262144);
    float*    yf  = (float*)   (smem + 263168);
    _Float16* yh  = (_Float16*)(smem + 265216);
    _Float16* yl  = (_Float16*)(smem + 266240);

    const int tid = threadIdx.x;
    const int c   = blockIdx.x;
    const float* Wc = W + (size_t)c * NGF;

    // ---- Setup: expand packed triangle into skew-symmetric S (f16 hi/lo split) ----
    // Diagonal of S is zero.
    Shi[tid * 256 + tid] = (_Float16)0.0f;
    Slo[tid * 256 + tid] = (_Float16)0.0f;

    // x column for this class; y0 = x (first Neumann iterate).
    {
        const float xv = x[(size_t)tid * NUM_CLS + c];
        xs[tid] = xv;
        yf[tid] = xv;
        const _Float16 xh = (_Float16)xv;
        yh[tid] = xh;
        yl[tid] = (_Float16)(xv - (float)xh);
    }

    // Thread-linear, coalesced sweep over the 32640 packed weights; scatter both
    // triangles of S into LDS.  idx -> (i, j) via inverse triangular numbering
    // (float estimate + exact integer fix-up).
    for (int t = tid; t < NGF; t += 256) {
        const float ft = (float)t;
        int i = (int)((511.0f - sqrtf(511.0f * 511.0f - 8.0f * ft)) * 0.5f);
        if (i < 0)   i = 0;
        if (i > 254) i = 254;
        while (i > 0 && trioff(i) > t) --i;
        while (trioff(i + 1) <= t)     ++i;
        const int j = i + 1 + (t - trioff(i));

        const float    w = 0.5f * Wc[t];
        const _Float16 h = (_Float16)w;
        const _Float16 l = (_Float16)(w - (float)h);
        Shi[i * 256 + j] =  h;  Slo[i * 256 + j] =  l;
        Shi[j * 256 + i] = -h;  Slo[j * 256 + i] = -l;
    }
    __syncthreads();

    // ---- Neumann iteration  y <- x - S*y  (converges: ||S|| ~ 0.32) ----
    const int lane  = tid & 31;
    const int wave  = tid >> 5;
    const int lhalf = lane >> 4;   // 0: lanes 0-15, 1: lanes 16-31
    const int lrow  = lane & 15;

    int rb = 0;                    // double-buffer index for y
    for (int it = 0; it < ITERS; ++it) {
        const int wb = rb ^ 1;
        const _Float16* yhr = yh + rb * 256;
        const _Float16* ylr = yl + rb * 256;

        for (int cb = 0; cb < 2; ++cb) {
            const int bi   = wave * 2 + cb;       // 16-row chunk [0, 16)
            const int arow = bi * 16 + lrow;      // row of S this lane supplies to A
            v8f acc = {};

#pragma unroll
            for (int t = 0; t < 8; ++t) {         // K = 256 in steps of 32
                const int k0  = t * 32;
                // A fragment (16x32 f16): lane holds row `arow`,
                // K runs {k0+0..7, k0+16..23} (lanes 0-15) or {k0+8..15, k0+24..31}.
                const int kA0 = k0 + (lhalf ? 8  : 0);
                const int kA1 = k0 + (lhalf ? 24 : 16);
                U16 Ah, Al, Bh, Bl;
                Ah.h[0] = *(const v8h*)(Shi + arow * 256 + kA0);
                Ah.h[1] = *(const v8h*)(Shi + arow * 256 + kA1);
                Al.h[0] = *(const v8h*)(Slo + arow * 256 + kA0);
                Al.h[1] = *(const v8h*)(Slo + arow * 256 + kA1);
                // B fragment (32x16 f16): every column = y chunk, value depends
                // only on K; K runs {k0+0..7, k0+16..23} (lanes 0-15) or {k0+8..15, k0+24..31}.
                const int kB = k0 + lhalf * 8;
                Bh.h[0] = *(const v8h*)(yhr + kB);
                Bh.h[1] = *(const v8h*)(yhr + kB + 16);
                Bl.h[0] = *(const v8h*)(ylr + kB);
                Bl.h[1] = *(const v8h*)(ylr + kB + 16);

                // Split-f16 product: S*y ~= Shi*yh + Slo*yh + Shi*yl (fp32 acc)
                acc = __builtin_amdgcn_wmma_f32_16x16x32_f16(
                          false, Ah.v, false, Bh.v, (short)0, acc, false, false);
                acc = __builtin_amdgcn_wmma_f32_16x16x32_f16(
                          false, Al.v, false, Bh.v, (short)0, acc, false, false);
                acc = __builtin_amdgcn_wmma_f32_16x16x32_f16(
                          false, Ah.v, false, Bl.v, (short)0, acc, false, false);
            }

            // D layout: element e, lanes 0-15 -> M=e (N=lane); lanes 16-31 -> M=8+e.
            // All 16 columns identical; lanes with N==0 commit the update.
            if (lrow == 0) {
#pragma unroll
                for (int e = 0; e < 8; ++e) {
                    const int   m    = bi * 16 + lhalf * 8 + e;
                    const float ynew = xs[m] - acc[e];
                    yf[wb * 256 + m] = ynew;
                    const _Float16 h = (_Float16)ynew;
                    yh[wb * 256 + m] = h;
                    yl[wb * 256 + m] = (_Float16)(ynew - (float)h);
                }
            }
        }
        __syncthreads();
        rb = wb;
    }

    // ---- out = (I - S)(I + S)^{-1} x = 2y - x ----
    out[(size_t)tid * NUM_CLS + c] = 2.0f * yf[rb * 256 + tid] - xs[tid];
}

extern "C" void kernel_launch(void* const* d_in, const int* in_sizes, int n_in,
                              void* d_out, int out_size, void* d_ws, size_t ws_size,
                              hipStream_t stream) {
    (void)in_sizes; (void)n_in; (void)d_ws; (void)ws_size; (void)out_size;
    const float* x = (const float*)d_in[0];   // (256, 512) fp32
    const float* W = (const float*)d_in[1];   // (512, 32640) fp32
    float* out = (float*)d_out;               // (256, 512) fp32

    // 256 KB of S tiles + small y buffers: needs the CDNA5 320 KB LDS cap.
    (void)hipFuncSetAttribute((const void*)cayley_rot_kernel,
                              hipFuncAttributeMaxDynamicSharedMemorySize,
                              SMEM_BYTES);
    cayley_rot_kernel<<<dim3(NUM_CLS), dim3(256), SMEM_BYTES, stream>>>(x, W, out);
}
